// QRNNLayer_48215302865708
// MI455X (gfx1250) — compile-verified, matrix operand order checked by reading, not run on hardware
//
#include <hip/hip_runtime.h>
#include <hip/hip_bf16.h>
#include <math.h>

// ---------------------------------------------------------------------------
// Bidirectional QRNN for MI455X (gfx1250, wave32, WMMA).
//
// gates GEMM:  M=32768 (B*L), N=1536 (2 dirs * 3 gates * 256), K=768 (3 taps * 256)
//              bf16 A/B, f32 accumulate, v_wmma_f32_16x16x32_bf16,
//              ping-pong register double-buffered K loop (no rotation movs).
// scan:        segmented parallel scan (16 segments x 256 steps), exact affine
//              recombination c_out = A*c_in + B with A = prod(f).
//
// Workspace layout (bytes), total ~221 MB:
//   xp    (bf16, halo-padded x) : off 0          size 16,785,408
//   wt    (bf16 fused weights)  : off 16,785,408 size  2,359,296
//   gates (f32)                 : off 19,144,704 size 201,326,592
//   segA/segB/cinit (f32)       : off 220,471,296 size 3*262,144
// ---------------------------------------------------------------------------

#define BATCH 8
#define SEQ   4096
#define LPAD  (SEQ + 2)
#define DIN   256
#define DOUT  256
#define NGATE 1536
#define KTOT  768
#define NSEG  16
#define SEGLEN 256
#define NCH   (BATCH * 2 * DOUT)   // 4096 scan channels

typedef __attribute__((ext_vector_type(16))) __bf16 v16bf;
typedef __attribute__((ext_vector_type(8)))  float  v8f;

union FragU { uint4 q[2]; v16bf v; };

__device__ __forceinline__ unsigned short f32_to_bf16(float f) {
  unsigned u = __float_as_uint(f);
  unsigned r = 0x7FFFu + ((u >> 16) & 1u);   // round-to-nearest-even
  return (unsigned short)((u + r) >> 16);
}

// -------------------- 1) x -> bf16 with zero halo rows ----------------------
__global__ void pad_convert(const float* __restrict__ x, unsigned short* __restrict__ xp) {
  int i = blockIdx.x * blockDim.x + threadIdx.x;
  if (i >= BATCH * LPAD * DIN) return;
  int c  = i & 255;
  int r  = i >> 8;
  int lp = r % LPAD;
  int b  = r / LPAD;
  float v = 0.f;
  if (lp >= 1 && lp <= SEQ)
    v = x[((size_t)(b * SEQ + (lp - 1)) << 8) + c];
  xp[i] = f32_to_bf16(v);
}

// -------------------- 2) fuse + transpose weights to [N][K] bf16 ------------
__global__ void build_wt(const float* __restrict__ Wf, const float* __restrict__ Wb,
                         unsigned short* __restrict__ wt) {
  int i = blockIdx.x * blockDim.x + threadIdx.x;
  if (i >= NGATE * KTOT) return;
  int k  = i % KTOT;
  int n  = i / KTOT;
  int dk = k >> 8;
  int c  = k & 255;
  float v = (n < 768) ? Wf[(size_t)n * 768 + c * 3 + dk]
                      : Wb[(size_t)(n - 768) * 768 + c * 3 + dk];
  wt[i] = f32_to_bf16(v);
}

// -------------------- 3) WMMA GEMM + bias + activation ----------------------
// Block: 256 threads = 8 waves; block tile 128Mx128N; wave tile 32Mx64N
// (2 A-frags x 4 B-frags -> 8 v_wmma_f32_16x16x32_bf16 per K-step of 32).

__device__ __forceinline__ void load_a_frags(const unsigned short* const abase[2],
                                             int k0, int lh, v16bf a[2]) {
  const int tap = k0 >> 8;                 // whole 32-wide K step stays in one tap
  const int cb  = (k0 & 255) + lh * 8;
#pragma unroll
  for (int mf = 0; mf < 2; ++mf) {
    const unsigned short* p = abase[mf] + (tap << 8) + cb;
    FragU u;
    u.q[0] = *(const uint4*)p;             // K = k0 + lh*8 .. +7
    u.q[1] = *(const uint4*)(p + 16);      // K = k0 + 16 + lh*8 .. +7
    a[mf] = u.v;
  }
}

__device__ __forceinline__ void load_b_frags(const unsigned short* const bbase[4],
                                             int k0, int lh, v16bf bm[4]) {
#pragma unroll
  for (int nf = 0; nf < 4; ++nf) {
    const unsigned short* p = bbase[nf] + k0 + lh * 8;
    FragU u;
    u.q[0] = *(const uint4*)p;
    u.q[1] = *(const uint4*)(p + 16);
    bm[nf] = u.v;
  }
}

__device__ __forceinline__ void wmma_tile(v8f acc[2][4], const v16bf a[2], const v16bf bm[4]) {
#pragma unroll
  for (int mf = 0; mf < 2; ++mf)
#pragma unroll
    for (int nf = 0; nf < 4; ++nf)
      acc[mf][nf] = __builtin_amdgcn_wmma_f32_16x16x32_bf16(
          false, a[mf], false, bm[nf], (short)0, acc[mf][nf], false, false);
}

__global__ void __launch_bounds__(256) gemm_gates_wmma(
    const unsigned short* __restrict__ xp,
    const unsigned short* __restrict__ wt,
    const float* __restrict__ bias_f,
    const float* __restrict__ bias_b,
    float* __restrict__ gates)
{
  const int lane = threadIdx.x & 31;
  const int wave = threadIdx.x >> 5;
  const int wm = wave & 3;              // 4 M slices of 32 rows
  const int wn = wave >> 2;             // 2 N slices of 64 cols
  const int ln = lane & 15;
  const int lh = lane >> 4;
  const int m0 = blockIdx.y * 128 + wm * 32;
  const int n0 = blockIdx.x * 128 + wn * 64;

  const v8f vzero = {0.f, 0.f, 0.f, 0.f, 0.f, 0.f, 0.f, 0.f};
  v8f acc[2][4];
#pragma unroll
  for (int mf = 0; mf < 2; ++mf)
#pragma unroll
    for (int nf = 0; nf < 4; ++nf) acc[mf][nf] = vzero;

  // Per-lane A row base (halo-padded, so no boundary branches in the K loop).
  const unsigned short* abase[2];
#pragma unroll
  for (int mf = 0; mf < 2; ++mf) {
    int r = m0 + mf * 16 + ln;          // global row = b*SEQ + l
    int b = r >> 12;
    int l = r & (SEQ - 1);
    abase[mf] = xp + ((size_t)(b * LPAD + l) << 8);
  }
  const unsigned short* bbase[4];
#pragma unroll
  for (int nf = 0; nf < 4; ++nf) {
    int n = n0 + nf * 16 + ln;
    bbase[nf] = wt + (size_t)n * KTOT;
  }

  // Software-pipelined, 2x-unrolled K loop with ping-pong fragment buffers:
  // loads for step k+32 are issued before the WMMAs of step k, and the
  // even/odd buffer roles alternate so no register-rotation movs are needed.
  v16bf a0[2], b0[4], a1[2], b1[4];
  load_a_frags(abase, 0, lh, a0);
  load_b_frags(bbase, 0, lh, b0);

#pragma unroll 1
  for (int k0 = 0; k0 < KTOT - 64; k0 += 64) {
    load_a_frags(abase, k0 + 32, lh, a1);
    load_b_frags(bbase, k0 + 32, lh, b1);
    wmma_tile(acc, a0, b0);
    load_a_frags(abase, k0 + 64, lh, a0);
    load_b_frags(bbase, k0 + 64, lh, b0);
    wmma_tile(acc, a1, b1);
  }
  // Peeled final pair (k0 = KTOT-64, KTOT-32).
  load_a_frags(abase, KTOT - 32, lh, a1);
  load_b_frags(bbase, KTOT - 32, lh, b1);
  wmma_tile(acc, a0, b0);
  wmma_tile(acc, a1, b1);

  // Epilogue: bias + branchless activation.
  // sigmoid(x) = rcp(1 + exp(-x));  tanh(x) = 2*sigmoid(2x) - 1.
#pragma unroll
  for (int nf = 0; nf < 4; ++nf) {
    const int col = n0 + nf * 16 + ln;
    const float bias = (col < 768) ? bias_f[col] : bias_b[col - 768];
    const int sub = (col < 768) ? col : col - 768;
    const bool is_tanh = (sub >= 512);
    const float pre_scale  = is_tanh ? 2.f : 1.f;
    const float post_scale = is_tanh ? 2.f : 1.f;
    const float post_off   = is_tanh ? -1.f : 0.f;
#pragma unroll
    for (int mf = 0; mf < 2; ++mf) {
#pragma unroll
      for (int v = 0; v < 8; ++v) {
        const int row = m0 + mf * 16 + (lh << 3) + v;   // C layout: M = v + 8*lh
        float xv = acc[mf][nf][v] + bias;
        float e  = __expf(-pre_scale * xv);             // v_exp_f32
        float s  = __builtin_amdgcn_rcpf(1.f + e);      // v_rcp_f32
        float g  = __builtin_fmaf(post_scale, s, post_off);
        gates[(size_t)row * NGATE + col] = g;
      }
    }
  }
}

// -------------------- 4) per-segment affine summaries -----------------------
// thread -> (b, dir, seg, d); d fastest for coalesced gate reads.
__global__ void scan_partial(const float* __restrict__ gates,
                             float* __restrict__ segA, float* __restrict__ segB) {
  int tid = blockIdx.x * blockDim.x + threadIdx.x;
  int d   = tid & 255;
  int t1  = tid >> 8;
  int seg = t1 & (NSEG - 1);
  int t2  = t1 >> 4;
  int dir = t2 & 1;
  int b   = t2 >> 1;
  size_t goff = (size_t)dir * 768 + d;

  float A = 1.f, c = 0.f;
  for (int s = 0; s < SEGLEN; ++s) {
    int sl = seg * SEGLEN + s;
    int t  = dir ? (SEQ - 1 - sl) : sl;
    const float* g = gates + (size_t)(b * SEQ + t) * NGATE + goff;
    float f = g[0];
    float z = g[512];
    A *= f;
    c = f * c + (1.f - f) * z;
  }
  int chan = (t2 << 8) + d;               // (b*2+dir)*256 + d
  segA[chan * NSEG + seg] = A;
  segB[chan * NSEG + seg] = c;
}

// -------------------- 5) serial combine across segments ---------------------
__global__ void scan_combine(const float* __restrict__ segA, const float* __restrict__ segB,
                             float* __restrict__ cinit) {
  int chan = blockIdx.x * blockDim.x + threadIdx.x;   // 4096
  float c = 0.f;
  for (int seg = 0; seg < NSEG; ++seg) {
    cinit[chan * NSEG + seg] = c;
    c = segA[chan * NSEG + seg] * c + segB[chan * NSEG + seg];
  }
}

// -------------------- 6) final scan with correct init + outputs -------------
__global__ void scan_final(const float* __restrict__ gates, const float* __restrict__ cinit,
                           float* __restrict__ out, float* __restrict__ last_h,
                           float* __restrict__ last_c) {
  int tid = blockIdx.x * blockDim.x + threadIdx.x;
  int d   = tid & 255;
  int t1  = tid >> 8;
  int seg = t1 & (NSEG - 1);
  int t2  = t1 >> 4;
  int dir = t2 & 1;
  int b   = t2 >> 1;
  size_t goff = (size_t)dir * 768 + d;
  int chan = (t2 << 8) + d;

  float c = cinit[chan * NSEG + seg];
  for (int s = 0; s < SEGLEN; ++s) {
    int sl = seg * SEGLEN + s;
    int t  = dir ? (SEQ - 1 - sl) : sl;
    const float* g = gates + (size_t)(b * SEQ + t) * NGATE + goff;
    float f = g[0];
    float o = g[256];
    float z = g[512];
    c = f * c + (1.f - f) * z;
    float h = c * o;
    out[(size_t)(b * SEQ + t) * 512 + dir * 256 + d] = h;
    if (t == SEQ - 1) {                // fwd: last step; bwd: first step
      last_h[b * 512 + dir * 256 + d] = h;
      last_c[b * 512 + dir * 256 + d] = c;
    }
  }
}

// ---------------------------------------------------------------------------
extern "C" void kernel_launch(void* const* d_in, const int* in_sizes, int n_in,
                              void* d_out, int out_size, void* d_ws, size_t ws_size,
                              hipStream_t stream) {
  const float* x   = (const float*)d_in[0];
  // d_in[1] = lengths: generated by the harness but unused by the reference math.
  const float* Wf  = (const float*)d_in[2];
  const float* bfw = (const float*)d_in[3];
  const float* Wb  = (const float*)d_in[4];
  const float* bbw = (const float*)d_in[5];

  unsigned char* ws = (unsigned char*)d_ws;
  const size_t OFF_XP    = 0;
  const size_t OFF_WT    = OFF_XP + (size_t)BATCH * LPAD * DIN * 2;        // 16,785,408
  const size_t OFF_GATES = OFF_WT + (size_t)NGATE * KTOT * 2;              // 19,144,704
  const size_t OFF_SEGA  = OFF_GATES + (size_t)BATCH * SEQ * NGATE * 4;    // 220,471,296
  unsigned short* xp  = (unsigned short*)(ws + OFF_XP);
  unsigned short* wt  = (unsigned short*)(ws + OFF_WT);
  float* gates        = (float*)(ws + OFF_GATES);
  float* segA         = (float*)(ws + OFF_SEGA);
  float* segB         = segA + (size_t)NCH * NSEG;
  float* cinit        = segB + (size_t)NCH * NSEG;

  float* out    = (float*)d_out;
  float* last_h = out + (size_t)BATCH * SEQ * 512;
  float* last_c = last_h + BATCH * 512;

  pad_convert<<<(BATCH * LPAD * DIN + 255) / 256, 256, 0, stream>>>(x, xp);
  build_wt<<<(NGATE * KTOT + 255) / 256, 256, 0, stream>>>(Wf, Wb, wt);
  gemm_gates_wmma<<<dim3(NGATE / 128, (BATCH * SEQ) / 128), 256, 0, stream>>>(
      xp, wt, bfw, bbw, gates);
  scan_partial<<<(NCH * NSEG) / 256, 256, 0, stream>>>(gates, segA, segB);
  scan_combine<<<NCH / 256, 256, 0, stream>>>(segA, segB, cinit);
  scan_final<<<(NCH * NSEG) / 256, 256, 0, stream>>>(gates, cinit, out, last_h, last_c);
}
